// BinaryModule_75788992905470
// MI455X (gfx1250) — matrix operand order repaired and checked by reading, not compiled, not run on hardware
//
#include <hip/hip_runtime.h>
#include <hip/hip_bf16.h>

// ---------------------------------------------------------------------------
// GAT network for 2048 graphs of 16 nodes / 128 edges, HEADS=2, C=256.
// Dominant cost: two [32768x256]@[256x512] GEMMs -> WMMA bf16 (f32 accum).
// Everything else is tiny per-graph work done one-block-per-graph in LDS.
// ---------------------------------------------------------------------------

#define NUM_GRAPHS      2048
#define NODES_PER_GRAPH 16
#define EDGES_PER_GRAPH 128
#define N_NODES         (NUM_GRAPHS * NODES_PER_GRAPH)   // 32768
#define N_EDGES         (NUM_GRAPHS * EDGES_PER_GRAPH)   // 262144
#define CH              256
#define MTOT            512                               // HEADS*CH
#define ETOT            (EDGES_PER_GRAPH + NODES_PER_GRAPH) // 144 incl self loops

typedef __attribute__((ext_vector_type(16))) __bf16 v16bf;
typedef __attribute__((ext_vector_type(8)))  __bf16 v8bf;
typedef __attribute__((ext_vector_type(8)))  float  v8f;

// ---------------------------------------------------------------------------
// f32 -> bf16 conversion (bandwidth pass)
// ---------------------------------------------------------------------------
__global__ void cvt_f32_to_bf16(const float* __restrict__ src,
                                __bf16* __restrict__ dst, int n) {
    int i = blockIdx.x * blockDim.x + threadIdx.x;
    if (i < n) dst[i] = (__bf16)src[i];
}

// ---------------------------------------------------------------------------
// D[N,512] = A[N,256] (row major, bf16) @ B[512,256]^T (B row major = W)
// One 16x16 f32 tile per wave32; K-loop 8 x v_wmma_f32_16x16x32_bf16.
// grid = (N/16, 4), block = 256 (8 waves; wave w -> colbase = by*128 + w*16)
// ---------------------------------------------------------------------------
__global__ void gemm_bf16_wmma(const __bf16* __restrict__ A,
                               const __bf16* __restrict__ B,
                               float* __restrict__ D) {
    const int lda = CH, ldb = CH, ldd = MTOT;
    const int wave = threadIdx.x >> 5;
    const int lane = threadIdx.x & 31;
    const int rowbase = blockIdx.x * 16;
    const int colbase = blockIdx.y * 128 + wave * 16;
    const int l15 = lane & 15;
    const bool hi = lane >= 16;

    // A fragment (16x32 bf16): lane = row M; lanes0-15 K={0..7,16..23},
    // lanes16-31 K={8..15,24..31} -> two contiguous 8-elem (16B) loads.
    const __bf16* arow = A + (size_t)(rowbase + l15) * lda;
    const int akoff = hi ? 8 : 0;
    // B fragment (32x16): lane = col N (lane&15); lanes0-15 K={0..7,16..23},
    // lanes16-31 K={8..15,24..31} -> two contiguous 8-elem (16B) loads.
    const __bf16* brow = B + (size_t)(colbase + l15) * ldb;
    const int bkoff = hi ? 8 : 0;

    v8f c = {};
#pragma unroll
    for (int k0 = 0; k0 < CH; k0 += 32) {
        v8bf a_lo = *(const v8bf*)(arow + k0 + akoff);
        v8bf a_hi = *(const v8bf*)(arow + k0 + akoff + 16);
        v8bf b_lo = *(const v8bf*)(brow + k0 + bkoff);
        v8bf b_hi = *(const v8bf*)(brow + k0 + bkoff + 16);
        v16bf a, b;
#pragma unroll
        for (int i = 0; i < 8; ++i) {
            a[i] = a_lo[i]; a[i + 8] = a_hi[i];
            b[i] = b_lo[i]; b[i + 8] = b_hi[i];
        }
        // (neg_a, A, neg_b, B, c_mod, C, reuse_a, reuse_b)
        c = __builtin_amdgcn_wmma_f32_16x16x32_bf16(false, a, false, b,
                                                    (short)0, c, false, false);
    }
    // D layout: VGPR r -> row = rowbase + r + (hi?8:0), col = colbase + (lane&15)
    float* drow = D + (size_t)(rowbase + (hi ? 8 : 0)) * ldd + colbase + l15;
#pragma unroll
    for (int r = 0; r < 8; ++r) drow[(size_t)r * ldd] = c[r];
}

// ---------------------------------------------------------------------------
// Per-graph GAT attention + aggregation. One block (256 thr) per graph.
// H: [N,512] f32 (head-major: h*256+c). Writes mean-over-heads + bias,
// optionally ReLU + bf16 (layer 1) or f32 (layer 2).
// ---------------------------------------------------------------------------
template <bool RELU_BF16>
__global__ void gat_layer(const float* __restrict__ H,
                          const int* __restrict__ esrc,
                          const int* __restrict__ edst,
                          const float* __restrict__ att_src,  // [2,256]
                          const float* __restrict__ att_dst,  // [2,256]
                          const float* __restrict__ bias,     // [256]
                          __bf16* __restrict__ out_bf,
                          float* __restrict__ out_f) {
    const int g = blockIdx.x;
    const int t = threadIdx.x;
    const int NODE0 = g * NODES_PER_GRAPH;

    __shared__ float s_as[16][2], s_ad[16][2];
    __shared__ int   s_s[ETOT], s_d[ETOT];
    __shared__ float s_w[ETOT][2];
    __shared__ float s_m[16][2], s_sum[16][2];
    __shared__ float red[256][2];

    // --- per-node attention dots: 32 (node,head) combos x 8 threads each ---
    {
        const int combo = t >> 3, sub = t & 7;
        const int n = combo >> 1, h = combo & 1;
        const float* hp = H + (size_t)(NODE0 + n) * MTOT + h * CH;
        const float* as = att_src + h * CH;
        const float* ad = att_dst + h * CH;
        float ps = 0.f, pd = 0.f;
        for (int c = sub * 32; c < sub * 32 + 32; ++c) {
            const float v = hp[c];
            ps += v * as[c];
            pd += v * ad[c];
        }
        red[t][0] = ps; red[t][1] = pd;
    }
    // --- edge lists (local indices) + self loops ---
    if (t < EDGES_PER_GRAPH) {
        s_s[t] = esrc[g * EDGES_PER_GRAPH + t] - NODE0;
        s_d[t] = edst[g * EDGES_PER_GRAPH + t] - NODE0;
    } else if (t < ETOT) {
        s_s[t] = t - EDGES_PER_GRAPH;
        s_d[t] = t - EDGES_PER_GRAPH;
    }
    __syncthreads();
    if (t < 32) {
        float ps = 0.f, pd = 0.f;
#pragma unroll
        for (int sub = 0; sub < 8; ++sub) {
            ps += red[t * 8 + sub][0];
            pd += red[t * 8 + sub][1];
        }
        s_as[t >> 1][t & 1] = ps;
        s_ad[t >> 1][t & 1] = pd;
    }
    __syncthreads();
    // --- leaky-relu logits per edge/head ---
    if (t < ETOT) {
#pragma unroll
        for (int h = 0; h < 2; ++h) {
            float a = s_as[s_s[t]][h] + s_ad[s_d[t]][h];
            s_w[t][h] = (a > 0.f) ? a : 0.2f * a;
        }
    }
    __syncthreads();
    // --- segment max / sum per (dst,head) ---
    if (t < 32) {
        const int n = t >> 1, h = t & 1;
        float m = -3.0e38f;
        for (int e = 0; e < ETOT; ++e)
            if (s_d[e] == n) m = fmaxf(m, s_w[e][h]);
        float s = 0.f;
        for (int e = 0; e < ETOT; ++e)
            if (s_d[e] == n) s += __expf(s_w[e][h] - m);
        s_m[n][h] = m;
        s_sum[n][h] = s + 1e-16f;
    }
    __syncthreads();
    if (t < ETOT) {
#pragma unroll
        for (int h = 0; h < 2; ++h)
            s_w[t][h] = __expf(s_w[t][h] - s_m[s_d[t]][h]) / s_sum[s_d[t]][h];
    }
    __syncthreads();
    // --- aggregation: thread t -> node t>>4, 16 channels starting (t&15)*16 ---
    {
        const int n = t >> 4;
        const int cb = (t & 15) * 16;
        float acc0[16], acc1[16];
#pragma unroll
        for (int c = 0; c < 16; ++c) { acc0[c] = 0.f; acc1[c] = 0.f; }
        for (int e = 0; e < ETOT; ++e) {
            if (s_d[e] != n) continue;
            const float w0 = s_w[e][0], w1 = s_w[e][1];
            const float* hs = H + (size_t)(NODE0 + s_s[e]) * MTOT + cb;
#pragma unroll
            for (int c = 0; c < 16; ++c) {
                acc0[c] += w0 * hs[c];
                acc1[c] += w1 * hs[CH + c];
            }
        }
        const size_t obase = (size_t)(NODE0 + n) * CH + cb;
#pragma unroll
        for (int c = 0; c < 16; ++c) {
            float v = 0.5f * (acc0[c] + acc1[c]) + bias[cb + c];
            if (RELU_BF16) {
                v = fmaxf(v, 0.f);
                out_bf[obase + c] = (__bf16)v;
            } else {
                out_f[obase + c] = v;
            }
        }
    }
}

// ---------------------------------------------------------------------------
// Per-graph attention pooling softmax + 2-class linear head.
// ---------------------------------------------------------------------------
__global__ void pool_classify(const float* __restrict__ h2,     // [N,256]
                              const float* __restrict__ w_att,  // [256]
                              const float* __restrict__ b_att,  // [1]
                              const float* __restrict__ Wc,     // [2,256]
                              const float* __restrict__ bc,     // [2]
                              float* __restrict__ out) {        // [G,2]
    const int g = blockIdx.x, t = threadIdx.x;
    __shared__ float red[256];
    __shared__ float s_logit[16];
    __shared__ float s_e[16];
    __shared__ float s_w[16];
    __shared__ float red2[256][2];

    // logits: 16 nodes x 16 threads each
    {
        const int n = t >> 4, sub = t & 15;
        const float* hp = h2 + (size_t)(g * 16 + n) * CH;
        float p = 0.f;
        for (int c = sub * 16; c < sub * 16 + 16; ++c) p += hp[c] * w_att[c];
        red[t] = p;
    }
    __syncthreads();
    if (t < 16) {
        float s = 0.f;
#pragma unroll
        for (int i = 0; i < 16; ++i) s += red[t * 16 + i];
        s_logit[t] = s + b_att[0];
    }
    __syncthreads();
    if (t < 16) {
        float m = -3e38f;
        for (int i = 0; i < 16; ++i) m = fmaxf(m, s_logit[i]);
        s_e[t] = __expf(s_logit[t] - m);
    }
    __syncthreads();
    if (t < 16) {
        float s = 0.f;
        for (int i = 0; i < 16; ++i) s += s_e[i];
        s_w[t] = s_e[t] / (s + 1e-16f);
    }
    __syncthreads();
    // pooled channel t
    float pooled = 0.f;
    for (int n = 0; n < 16; ++n)
        pooled += s_w[n] * h2[(size_t)(g * 16 + n) * CH + t];
    red2[t][0] = pooled * Wc[t];
    red2[t][1] = pooled * Wc[CH + t];
    __syncthreads();
    for (int stride = 128; stride > 0; stride >>= 1) {
        if (t < stride) {
            red2[t][0] += red2[t + stride][0];
            red2[t][1] += red2[t + stride][1];
        }
        __syncthreads();
    }
    if (t == 0) {
        out[g * 2 + 0] = red2[0][0] + bc[0];
        out[g * 2 + 1] = red2[0][1] + bc[1];
    }
}

// ---------------------------------------------------------------------------
extern "C" void kernel_launch(void* const* d_in, const int* in_sizes, int n_in,
                              void* d_out, int out_size, void* d_ws, size_t ws_size,
                              hipStream_t stream) {
    (void)in_sizes; (void)n_in; (void)out_size; (void)ws_size;

    const float* x     = (const float*)d_in[0];
    const int*   eidx  = (const int*)d_in[1];
    // d_in[2] = batch (unused: nodes are contiguous per graph by construction)
    const float* W1    = (const float*)d_in[3];
    const float* as1   = (const float*)d_in[4];
    const float* ad1   = (const float*)d_in[5];
    const float* b1    = (const float*)d_in[6];
    const float* W2    = (const float*)d_in[7];
    const float* as2   = (const float*)d_in[8];
    const float* ad2   = (const float*)d_in[9];
    const float* b2    = (const float*)d_in[10];
    const float* w_att = (const float*)d_in[11];
    const float* b_att = (const float*)d_in[12];
    const float* Wc    = (const float*)d_in[13];
    const float* bc    = (const float*)d_in[14];
    float* out = (float*)d_out;

    // Workspace layout (~128.5 MB). Hbig reused for both GEMM outputs.
    char* ws = (char*)d_ws;
    size_t off = 0;
    float*  Hbig = (float*)(ws + off);  off += (size_t)N_NODES * MTOT * 4;  // 64MB
    float*  h2   = (float*)(ws + off);  off += (size_t)N_NODES * CH * 4;    // 32MB
    __bf16* xb   = (__bf16*)(ws + off); off += (size_t)N_NODES * CH * 2;    // 16MB
    __bf16* h1b  = (__bf16*)(ws + off); off += (size_t)N_NODES * CH * 2;    // 16MB
    __bf16* W1b  = (__bf16*)(ws + off); off += (size_t)MTOT * CH * 2;
    __bf16* W2b  = (__bf16*)(ws + off); off += (size_t)MTOT * CH * 2;

    const int* esrc = eidx;
    const int* edst = eidx + N_EDGES;

    const int nx = N_NODES * CH;
    const int nw = MTOT * CH;
    cvt_f32_to_bf16<<<(nx + 255) / 256, 256, 0, stream>>>(x, xb, nx);
    cvt_f32_to_bf16<<<(nw + 255) / 256, 256, 0, stream>>>(W1, W1b, nw);
    cvt_f32_to_bf16<<<(nw + 255) / 256, 256, 0, stream>>>(W2, W2b, nw);

    dim3 ggrid(N_NODES / 16, MTOT / 128);
    gemm_bf16_wmma<<<ggrid, 256, 0, stream>>>(xb, W1b, Hbig);
    gat_layer<true><<<NUM_GRAPHS, 256, 0, stream>>>(Hbig, esrc, edst, as1, ad1,
                                                    b1, h1b, nullptr);
    gemm_bf16_wmma<<<ggrid, 256, 0, stream>>>(h1b, W2b, Hbig);
    gat_layer<false><<<NUM_GRAPHS, 256, 0, stream>>>(Hbig, esrc, edst, as2, ad2,
                                                     b2, nullptr, h2);
    pool_classify<<<NUM_GRAPHS, 256, 0, stream>>>(h2, w_att, b_att, Wc, bc, out);
}
